// MambaEncoder_53850299957260
// MI455X (gfx1250) — matrix operand order repaired and checked by reading, not compiled, not run on hardware
//
#include <hip/hip_runtime.h>
#include <cstdint>
#include <cstddef>

// ---------------------------------------------------------------------------
// Mamba encoder for MI455X (gfx1250, wave32, WMMA).
// GEMMs: f16 via v_wmma_f32_16x16x32_f16, f32 accumulate, double-buffered LDS
// with async global->LDS copies (ASYNCcnt) when the toolchain exposes them.
// Selective scan: chunked two-pass associative scan for 16x wave parallelism.
// ---------------------------------------------------------------------------

constexpr int S_LEN   = 4096;
constexpr int D_MODEL = 1024;
constexpr int D_INNER = 2048;
constexpr int N_STATE = 16;
constexpr int DT_RANK = 64;
constexpr int N_DBC   = 96;   // DT_RANK + 2*N_STATE
constexpr int N_LAYERS = 4;
constexpr int NCHUNK  = 16;               // scan chunks
constexpr int CHUNK   = S_LEN / NCHUNK;   // 256
constexpr int SCN     = D_INNER * N_STATE; // 32768 scan lanes

typedef __attribute__((ext_vector_type(16))) _Float16 v16h;
typedef __attribute__((ext_vector_type(8)))  _Float16 v8h;
typedef __attribute__((ext_vector_type(8)))  float    v8f;

union Frag { v16h v; struct { v8h lo, hi; } h; };

// --- async global->LDS copies (guarded; fallback = reg-staged sync copy) ---
#if __has_builtin(__builtin_amdgcn_global_load_async_to_lds_b128) && \
    __has_builtin(__builtin_amdgcn_s_wait_asynccnt)
#define MAMBA_ASYNC 1
// Signature (from clang diagnostic): param0 = AS1 ptr to int vector_size(16),
// param1 = AS3 ptr, param2/3 = imm offset / cpol.
typedef int v4i_vs __attribute__((vector_size(16)));
typedef __attribute__((address_space(1))) v4i_vs as1_v4i;
typedef __attribute__((address_space(3))) v4i_vs as3_v4i;
__device__ __forceinline__ void async_cp16(const void* g, void* l) {
  // generic LDS pointer: low 32 bits are the LDS offset (ISA 10.2 aperture)
  __builtin_amdgcn_global_load_async_to_lds_b128(
      (as1_v4i*)(uintptr_t)g, (as3_v4i*)(uint32_t)(uintptr_t)l, 0, 0);
}
#define AWAIT() __builtin_amdgcn_s_wait_asynccnt(0)
#else
#define MAMBA_ASYNC 0
#define AWAIT() ((void)0)
#endif

// ---------------------------------------------------------------------------
// Block-wide (256 thread) pairwise reduction of two floats.
// ---------------------------------------------------------------------------
__device__ __forceinline__ float2 block_reduce_2(float a, float b) {
  __shared__ float sa[256];
  __shared__ float sb[256];
  const int t = threadIdx.x;
  sa[t] = a; sb[t] = b;
  __syncthreads();
  for (int off = 128; off > 0; off >>= 1) {
    if (t < off) { sa[t] += sa[t + off]; sb[t] += sb[t + off]; }
    __syncthreads();
  }
  return make_float2(sa[0], sb[0]);
}

// ---------------------------------------------------------------------------
// enc = LayerNorm(x + pos_enc), eps = 1e-6, f32 out
// ---------------------------------------------------------------------------
__global__ __launch_bounds__(256)
void k_add_ln(const float* __restrict__ x, const float* __restrict__ pos,
              const float* __restrict__ g, const float* __restrict__ b,
              float* __restrict__ out, int D, float eps) {
  const int row = blockIdx.x;
  const float* xr = x + (size_t)row * D;
  const float* pr = pos + (size_t)row * D;
  float s = 0.f, q = 0.f;
  for (int c = threadIdx.x; c < D; c += 256) {
    float v = xr[c] + pr[c];
    s += v; q += v * v;
  }
  float2 r = block_reduce_2(s, q);
  float mu = r.x / D;
  float rs = rsqrtf(r.y / D - mu * mu + eps);
  float* orow = out + (size_t)row * D;
  for (int c = threadIdx.x; c < D; c += 256)
    orow[c] = (xr[c] + pr[c] - mu) * rs * g[c] + b[c];
}

// ---------------------------------------------------------------------------
// LayerNorm f32 -> f16 (feeds the WMMA GEMM A operand)
// ---------------------------------------------------------------------------
__global__ __launch_bounds__(256)
void k_ln_f16(const float* __restrict__ in, const float* __restrict__ g,
              const float* __restrict__ b, _Float16* __restrict__ out,
              int D, float eps) {
  const int row = blockIdx.x;
  const float* xr = in + (size_t)row * D;
  float s = 0.f, q = 0.f;
  for (int c = threadIdx.x; c < D; c += 256) { float v = xr[c]; s += v; q += v * v; }
  float2 r = block_reduce_2(s, q);
  float mu = r.x / D;
  float rs = rsqrtf(r.y / D - mu * mu + eps);
  _Float16* orow = out + (size_t)row * D;
  for (int c = threadIdx.x; c < D; c += 256)
    orow[c] = (_Float16)((xr[c] - mu) * rs * g[c] + b[c]);
}

// ---------------------------------------------------------------------------
// f32 -> f16 weight conversion
// ---------------------------------------------------------------------------
__global__ __launch_bounds__(256)
void k_cvt(const float* __restrict__ src, _Float16* __restrict__ dst, int n) {
  int i = blockIdx.x * 256 + threadIdx.x;
  if (i < n) dst[i] = (_Float16)src[i];
}

// dbc[:, 0:64] -> f16 (A operand of dt_proj GEMM)
__global__ __launch_bounds__(256)
void k_slice16(const float* __restrict__ dbc, _Float16* __restrict__ dst) {
  int i = blockIdx.x * 256 + threadIdx.x;
  int row = i >> 6;
  int j = i & 63;
  dst[i] = (_Float16)dbc[(size_t)row * N_DBC + j];
}

// ---------------------------------------------------------------------------
// WMMA GEMM: C[M,N] = A16[M,K] * Brow16[N,K]^T (+epilogue)
//   EPI 0: + bias (optional)
//   EPI 1: softplus(acc + bias)
//   EPI 2: + bias + skip[M,N]
// Block: 256 threads = 8 waves; block tile 128x128, BK=32; double-buffered
// LDS, async global->LDS copies overlap the v_wmma stream.
// Wave tile 32x64 -> 2x4 accumulators of 16x16 (v8f each).
// ---------------------------------------------------------------------------
template <int EPI>
__global__ __launch_bounds__(256)
void k_gemm(const _Float16* __restrict__ A, const _Float16* __restrict__ B,
            const float* __restrict__ bias, const float* __restrict__ skip,
            float* __restrict__ C, int M, int N, int K) {
  constexpr int BM = 128, BN = 128, BK = 32, SA = BK + 8;  // padded f16 stride
  __shared__ _Float16 As[2][BM * SA];
  __shared__ _Float16 Bs[2][BN * SA];

  const int tid  = threadIdx.x;
  const int lane = tid & 31;
  const int wid  = tid >> 5;
  const int wm   = (wid & 3) * 32;   // wave row offset in block tile
  const int wn   = (wid >> 2) * 64;  // wave col offset in block tile
  const int m0   = blockIdx.y * BM;
  const int n0   = blockIdx.x * BN;
  const int half = lane >> 4;        // lane half (0/1)
  const int lr   = lane & 15;

  v8f acc[2][4] = {};
  const int nTiles = K / BK;

  // Stage one BKxBM / BKxBN tile pair into LDS buffer `buf`.
  auto stage = [&](int k0, int buf) {
#pragma unroll
    for (int i = 0; i < 2; ++i) {
      int idx = tid + i * 256;
      int r   = idx >> 2;
      int kc  = (idx & 3) * 8;
      const _Float16* ga = A + (size_t)(m0 + r) * K + k0 + kc;
      int nr = n0 + r; if (nr >= N) nr = N - 1;  // clamp: dup rows feed only
      const _Float16* gb = B + (size_t)nr * K + k0 + kc;  // discarded cols
#if MAMBA_ASYNC
      async_cp16(ga, &As[buf][r * SA + kc]);
      async_cp16(gb, &Bs[buf][r * SA + kc]);
#else
      uint4 va = *reinterpret_cast<const uint4*>(ga);
      uint4 vb = *reinterpret_cast<const uint4*>(gb);
      *reinterpret_cast<uint4*>(&As[buf][r * SA + kc]) = va;
      *reinterpret_cast<uint4*>(&Bs[buf][r * SA + kc]) = vb;
#endif
    }
  };

  stage(0, 0);
  AWAIT();
  __syncthreads();

  for (int t = 0; t < nTiles; ++t) {
    const int cur = t & 1;
    if (t + 1 < nTiles) stage((t + 1) * BK, cur ^ 1);  // overlap with compute

    // --- build fragments per ISA 7.12.2 (wave32) ---
    Frag fa[2], fb[4];
#pragma unroll
    for (int tm = 0; tm < 2; ++tm) {
      // A: lane half 0 -> K[0..8)+[16..24); half 1 -> K[8..16)+[24..32)
      const _Float16* p = &As[cur][(wm + tm * 16 + lr) * SA + half * 8];
      fa[tm].h.lo = *reinterpret_cast<const v8h*>(p);
      fa[tm].h.hi = *reinterpret_cast<const v8h*>(p + 16);
    }
#pragma unroll
    for (int tn = 0; tn < 4; ++tn) {
      // B: lane half 0 -> K[0..8)+[16..24); half 1 -> K[8..16)+[24..32)
      const _Float16* p = &Bs[cur][(wn + tn * 16 + lr) * SA + half * 8];
      fb[tn].h.lo = *reinterpret_cast<const v8h*>(p);
      fb[tn].h.hi = *reinterpret_cast<const v8h*>(p + 16);
    }
#pragma unroll
    for (int tm = 0; tm < 2; ++tm)
#pragma unroll
      for (int tn = 0; tn < 4; ++tn)
        acc[tm][tn] = __builtin_amdgcn_wmma_f32_16x16x32_f16(
            false, fa[tm].v, false, fb[tn].v, (short)0, acc[tm][tn],
            false, false);

    AWAIT();           // my async copies into buf cur^1 landed
    __syncthreads();   // everyone's copies landed / reads of cur finished
  }

  // --- epilogue: C layout lane(0-15)->N, VGPR v -> M = half*8+v ---
#pragma unroll
  for (int tm = 0; tm < 2; ++tm)
#pragma unroll
    for (int tn = 0; tn < 4; ++tn) {
      int col = n0 + wn + tn * 16 + lr;
      if (col >= N) continue;
      float bval = bias ? bias[col] : 0.0f;
#pragma unroll
      for (int v = 0; v < 8; ++v) {
        int row = m0 + wm + tm * 16 + half * 8 + v;
        float xv = acc[tm][tn][v] + bval;
        if (EPI == 1) {
          xv = (xv > 20.0f) ? xv : log1pf(__expf(xv));
        } else if (EPI == 2) {
          xv += skip[(size_t)row * N + col];
        }
        C[(size_t)row * N + col] = xv;
      }
    }
}

// ---------------------------------------------------------------------------
// Depthwise conv1d k=3, pad=1 over sequence dim on x1 = xz[:, :D_INNER].
// Writes u (f32, for scan) and u16 (f16, for deltaBC GEMM).
// ---------------------------------------------------------------------------
__global__ __launch_bounds__(256)
void k_conv(const float* __restrict__ xz, const float* __restrict__ w,
            const float* __restrict__ bias, float* __restrict__ u,
            _Float16* __restrict__ u16) {
  int idx = blockIdx.x * 256 + threadIdx.x;        // idx = s*D_INNER + c
  int sRow = idx / D_INNER;
  int c = idx - sRow * D_INNER;
  float acc = bias[c];
  const float* wc = w + c * 3;
#pragma unroll
  for (int j = 0; j < 3; ++j) {
    int t = sRow - 1 + j;
    if (t >= 0 && t < S_LEN)
      acc += wc[j] * xz[(size_t)t * (2 * D_INNER) + c];
  }
  u[idx]   = acc;
  u16[idx] = (_Float16)acc;
}

// ---------------------------------------------------------------------------
// Chunked selective scan. Recurrence h' = a*h + b is associative, so:
//  pass1: per chunk j, per (c,n): P = prod(a), H = state from h_in=0
//  carry: chain 16 chunk boundaries sequentially (tiny)
//  pass2: replay each chunk from carry-in, producing y
// dbc cols: [0,64)=dt_r, [64,80)=B, [80,96)=C.
// ---------------------------------------------------------------------------
__global__ __launch_bounds__(256)
void k_scan_pass1(const float* __restrict__ u, const float* __restrict__ delta,
                  const float* __restrict__ dbc, const float* __restrict__ A_log,
                  float* __restrict__ P, float* __restrict__ H) {
  const int c = blockIdx.x * 256 + threadIdx.x;   // channel
  const int j = blockIdx.y;                        // chunk
  float Av[N_STATE], hs[N_STATE], pr[N_STATE];
#pragma unroll
  for (int n = 0; n < N_STATE; ++n) {
    Av[n] = -__expf(A_log[(size_t)c * N_STATE + n]);
    hs[n] = 0.f; pr[n] = 1.f;
  }
  const int t0 = j * CHUNK;
  for (int t = t0; t < t0 + CHUNK; ++t) {
    float dt = delta[(size_t)t * D_INNER + c];
    float uu = u[(size_t)t * D_INNER + c];
    float du = dt * uu;
    const float* bc = dbc + (size_t)t * N_DBC;
#pragma unroll
    for (int n = 0; n < N_STATE; ++n) {
      float a = __expf(dt * Av[n]);
      pr[n] *= a;
      hs[n] = a * hs[n] + du * bc[DT_RANK + n];
    }
  }
  const size_t base = (size_t)j * SCN + (size_t)c * N_STATE;
#pragma unroll
  for (int n = 0; n < N_STATE; ++n) { P[base + n] = pr[n]; H[base + n] = hs[n]; }
}

__global__ __launch_bounds__(256)
void k_scan_carry(const float* __restrict__ P, const float* __restrict__ H,
                  float* __restrict__ carry) {
  const int i = blockIdx.x * 256 + threadIdx.x;   // (c,n) flat, 32768 lanes
  float h = 0.f;
  for (int j = 0; j < NCHUNK; ++j) {
    carry[(size_t)j * SCN + i] = h;               // h at chunk start
    h = P[(size_t)j * SCN + i] * h + H[(size_t)j * SCN + i];
  }
}

__global__ __launch_bounds__(256)
void k_scan_pass2(const float* __restrict__ u, const float* __restrict__ delta,
                  const float* __restrict__ dbc, const float* __restrict__ A_log,
                  const float* __restrict__ Dp, const float* __restrict__ carry,
                  float* __restrict__ y) {
  const int c = blockIdx.x * 256 + threadIdx.x;
  const int j = blockIdx.y;
  float Av[N_STATE], hs[N_STATE];
  const size_t base = (size_t)j * SCN + (size_t)c * N_STATE;
#pragma unroll
  for (int n = 0; n < N_STATE; ++n) {
    Av[n] = -__expf(A_log[(size_t)c * N_STATE + n]);
    hs[n] = carry[base + n];
  }
  const float Dv = Dp[c];
  const int t0 = j * CHUNK;
  for (int t = t0; t < t0 + CHUNK; ++t) {
    float dt = delta[(size_t)t * D_INNER + c];
    float uu = u[(size_t)t * D_INNER + c];
    float du = dt * uu;
    const float* bc = dbc + (size_t)t * N_DBC;
    float acc = 0.f;
#pragma unroll
    for (int n = 0; n < N_STATE; ++n) {
      float h = __expf(dt * Av[n]) * hs[n] + du * bc[DT_RANK + n];
      acc += h * bc[DT_RANK + N_STATE + n];
      hs[n] = h;
    }
    y[(size_t)t * D_INNER + c] = acc + uu * Dv;
  }
}

// ---------------------------------------------------------------------------
// g16 = f16( LN(y; outnorm) * silu(z1) ), z1 = xz[:, D_INNER:2*D_INNER]
// ---------------------------------------------------------------------------
__global__ __launch_bounds__(256)
void k_gate(const float* __restrict__ y, const float* __restrict__ xz,
            const float* __restrict__ g, const float* __restrict__ b,
            _Float16* __restrict__ out, float eps) {
  const int row = blockIdx.x;
  const float* yr = y + (size_t)row * D_INNER;
  float s = 0.f, q = 0.f;
  for (int c = threadIdx.x; c < D_INNER; c += 256) {
    float v = yr[c]; s += v; q += v * v;
  }
  float2 r = block_reduce_2(s, q);
  float mu = r.x / D_INNER;
  float rs = rsqrtf(r.y / D_INNER - mu * mu + eps);
  const float* zr = xz + (size_t)row * (2 * D_INNER) + D_INNER;
  _Float16* orow = out + (size_t)row * D_INNER;
  for (int c = threadIdx.x; c < D_INNER; c += 256) {
    float ln  = (yr[c] - mu) * rs * g[c] + b[c];
    float z   = zr[c];
    float sil = z / (1.0f + __expf(-z));
    orow[c] = (_Float16)(ln * sil);
  }
}

// ---------------------------------------------------------------------------
// Host orchestration
// ---------------------------------------------------------------------------
extern "C" void kernel_launch(void* const* d_in, const int* in_sizes, int n_in,
                              void* d_out, int out_size, void* d_ws, size_t ws_size,
                              hipStream_t stream) {
  const float* x         = (const float*)d_in[0];
  const float* pos       = (const float*)d_in[1];
  const float* ln_g      = (const float*)d_in[2];
  const float* ln_b      = (const float*)d_in[3];
  const float* innorm_g  = (const float*)d_in[4];
  const float* innorm_b  = (const float*)d_in[5];
  const float* in_proj_w = (const float*)d_in[6];
  const float* in_proj_b = (const float*)d_in[7];
  const float* conv_w    = (const float*)d_in[8];
  const float* conv_b    = (const float*)d_in[9];
  const float* deltaBC_w = (const float*)d_in[10];
  const float* dt_proj_w = (const float*)d_in[11];
  const float* dt_proj_b = (const float*)d_in[12];
  const float* A_log     = (const float*)d_in[13];
  const float* Dp        = (const float*)d_in[14];
  const float* outnorm_g = (const float*)d_in[15];
  const float* outnorm_b = (const float*)d_in[16];
  const float* out_proj_w = (const float*)d_in[17];
  const float* out_proj_b = (const float*)d_in[18];

  char* p = (char*)d_ws;
  auto carve = [&p](size_t bytes) -> void* {
    void* r = (void*)p;
    p += (bytes + 255) & ~(size_t)255;
    return r;
  };
  float*    enc0  = (float*)carve((size_t)S_LEN * D_MODEL * 4);
  float*    enc1  = (float*)carve((size_t)S_LEN * D_MODEL * 4);
  _Float16* h16   = (_Float16*)carve((size_t)S_LEN * D_MODEL * 2);
  float*    xz    = (float*)carve((size_t)S_LEN * 2 * D_INNER * 4);
  float*    u     = (float*)carve((size_t)S_LEN * D_INNER * 4);
  _Float16* u16   = (_Float16*)carve((size_t)S_LEN * D_INNER * 2);
  float*    dbc   = (float*)carve((size_t)S_LEN * N_DBC * 4);
  _Float16* ds16  = (_Float16*)carve((size_t)S_LEN * DT_RANK * 2);
  float*    delta = (float*)carve((size_t)S_LEN * D_INNER * 4);
  float*    ybuf  = (float*)carve((size_t)S_LEN * D_INNER * 4);
  _Float16* g16   = (_Float16*)carve((size_t)S_LEN * D_INNER * 2);
  _Float16* wIN   = (_Float16*)carve((size_t)2 * D_INNER * D_MODEL * 2);
  _Float16* wDBC  = (_Float16*)carve((size_t)N_DBC * D_INNER * 2);
  _Float16* wDT   = (_Float16*)carve((size_t)D_INNER * DT_RANK * 2);
  _Float16* wOUT  = (_Float16*)carve((size_t)D_MODEL * D_INNER * 2);
  float*    scanP = (float*)carve((size_t)NCHUNK * SCN * 4);
  float*    scanH = (float*)carve((size_t)NCHUNK * SCN * 4);
  float*    scanC = (float*)carve((size_t)NCHUNK * SCN * 4);

  k_add_ln<<<S_LEN, 256, 0, stream>>>(x, pos, ln_g, ln_b, enc0, D_MODEL, 1e-6f);

  float* bufs[2] = {enc0, enc1};
  for (int l = 0; l < N_LAYERS; ++l) {
    float* encIn  = bufs[l & 1];
    float* encOut = (l == N_LAYERS - 1) ? (float*)d_out : bufs[(l + 1) & 1];

    const int nIN  = 2 * D_INNER * D_MODEL;
    const int nDBC = N_DBC * D_INNER;
    const int nDT  = D_INNER * DT_RANK;
    const int nOUT = D_MODEL * D_INNER;
    k_cvt<<<nIN / 256, 256, 0, stream>>>(in_proj_w + (size_t)l * nIN, wIN, nIN);
    k_cvt<<<nDBC / 256, 256, 0, stream>>>(deltaBC_w + (size_t)l * nDBC, wDBC, nDBC);
    k_cvt<<<nDT / 256, 256, 0, stream>>>(dt_proj_w + (size_t)l * nDT, wDT, nDT);
    k_cvt<<<nOUT / 256, 256, 0, stream>>>(out_proj_w + (size_t)l * nOUT, wOUT, nOUT);

    k_ln_f16<<<S_LEN, 256, 0, stream>>>(encIn, innorm_g + l * D_MODEL,
                                        innorm_b + l * D_MODEL, h16, D_MODEL, 1e-5f);

    dim3 g1((2 * D_INNER) / 128, S_LEN / 128);
    k_gemm<0><<<g1, 256, 0, stream>>>(h16, wIN, in_proj_b + (size_t)l * 2 * D_INNER,
                                      nullptr, xz, S_LEN, 2 * D_INNER, D_MODEL);

    k_conv<<<(S_LEN * D_INNER) / 256, 256, 0, stream>>>(
        xz, conv_w + (size_t)l * D_INNER * 3, conv_b + (size_t)l * D_INNER, u, u16);

    dim3 g2((N_DBC + 127) / 128, S_LEN / 128);
    k_gemm<0><<<g2, 256, 0, stream>>>(u16, wDBC, nullptr, nullptr, dbc,
                                      S_LEN, N_DBC, D_INNER);

    k_slice16<<<(S_LEN * DT_RANK) / 256, 256, 0, stream>>>(dbc, ds16);

    dim3 g3(D_INNER / 128, S_LEN / 128);
    k_gemm<1><<<g3, 256, 0, stream>>>(ds16, wDT, dt_proj_b + (size_t)l * D_INNER,
                                      nullptr, delta, S_LEN, D_INNER, DT_RANK);

    const float* Al = A_log + (size_t)l * D_INNER * N_STATE;
    dim3 gs(D_INNER / 256, NCHUNK);
    k_scan_pass1<<<gs, 256, 0, stream>>>(u, delta, dbc, Al, scanP, scanH);
    k_scan_carry<<<SCN / 256, 256, 0, stream>>>(scanP, scanH, scanC);
    k_scan_pass2<<<gs, 256, 0, stream>>>(u, delta, dbc, Al,
                                         Dp + (size_t)l * D_INNER, scanC, ybuf);

    k_gate<<<S_LEN, 256, 0, stream>>>(ybuf, xz, outnorm_g + (size_t)l * D_INNER,
                                      outnorm_b + (size_t)l * D_INNER, g16, 1e-5f);

    dim3 g4(D_MODEL / 128, S_LEN / 128);
    k_gemm<2><<<g4, 256, 0, stream>>>(g16, wOUT, out_proj_b + (size_t)l * D_MODEL,
                                      encIn, encOut, S_LEN, D_MODEL, D_INNER);
  }
  (void)in_sizes; (void)n_in; (void)out_size; (void)ws_size;
}
